// IterSpatialCorrelationSampler_21560735826388
// MI455X (gfx1250) — matrix operand order repaired and checked by reading, not compiled, not run on hardware
//
#include <hip/hip_runtime.h>
#include <hip/hip_bf16.h>

// Spatial correlation (PATCH=9, dil=1): out[b,di,dj,h,w] = sum_c in1[b,c,h,w]*in2pad[b,c,h+di,w+dj]
// Band-GEMM on v_wmma_f32_16x16x32_bf16 with fp32 = bf16_hi + bf16_lo split (3-term product).

#define MAXD 4
#define B_   8
#define C_   256
#define H_   96
#define W_   192
#define HW_  (H_ * W_)
#define CHW_ (C_ * HW_)

typedef __attribute__((ext_vector_type(16))) __bf16        v16bf;
typedef __attribute__((ext_vector_type(8)))  float         v8f;
typedef __attribute__((ext_vector_type(8)))  unsigned int  v8u;

// LDS strides padded so lanes 0-15 vs 16-31 of a fragment load hit disjoint bank ranges
#define A_S 20   // A pair-major [p(16)][m(16)]
#define B_S 28   // B pair-major [di*16+p][n(24)]

__device__ __forceinline__ unsigned short bf16_bits(float x) {
  return __builtin_bit_cast(unsigned short, (__bf16)x);  // fptrunc f32->bf16 (RNE)
}
// Pack the (k, k+1) channel pair at one column into hi-word / lo-word (Dekker split).
__device__ __forceinline__ void split_pack(float xk, float xk1, unsigned& hw, unsigned& lw) {
  unsigned short h0 = bf16_bits(xk);
  unsigned short h1 = bf16_bits(xk1);
  hw = ((unsigned)h1 << 16) | h0;
  float r0 = xk  - __uint_as_float((unsigned)h0 << 16);
  float r1 = xk1 - __uint_as_float((unsigned)h1 << 16);
  lw = ((unsigned)bf16_bits(r1) << 16) | bf16_bits(r0);
}

// A fragment (16x32 bf16): lane m holds K pairs; p = v + (hi-half?8:0)
__device__ __forceinline__ v16bf load_a_frag(const unsigned* s, int m, int sel) {
  v8u r;
#pragma unroll
  for (int v = 0; v < 8; ++v) {
    int p = v + (sel ? 8 : 0);
    r[v] = s[p * A_S + m];
  }
  return __builtin_bit_cast(v16bf, r);
}
// B fragment (32x16 bf16): lane n holds K pairs; p = v + (hi-half?8:0)
__device__ __forceinline__ v16bf load_b_frag(const unsigned* s, int base, int n, int sel) {
  v8u r;
#pragma unroll
  for (int v = 0; v < 8; ++v) {
    int p = v + (sel ? 8 : 0);
    r[v] = s[base + p * B_S + n];
  }
  return __builtin_bit_cast(v16bf, r);
}

__global__ __launch_bounds__(288) void IterSpatialCorrelationSampler_21560735826388_kernel(
    const float* __restrict__ in1, const float* __restrict__ in2, float* __restrict__ out) {
  __shared__ __attribute__((aligned(16))) unsigned s1h[16 * A_S];
  __shared__ __attribute__((aligned(16))) unsigned s1l[16 * A_S];
  __shared__ __attribute__((aligned(16))) unsigned s2h[9 * 16 * B_S];
  __shared__ __attribute__((aligned(16))) unsigned s2l[9 * 16 * B_S];
  __shared__ __attribute__((aligned(16))) float    scr[9 * 512];

  const int w0   = blockIdx.x * 16;
  const int h    = blockIdx.y;
  const int b    = blockIdx.z;
  const int tid  = threadIdx.x;
  const int lane = tid & 31;
  const int wv   = tid >> 5;       // wave id == di (0..8)
  const int mn   = lane & 15;      // M for A / N for B
  const int sel  = lane >> 4;      // half-wave select

  const float* in1b = in1 + (size_t)b * CHW_ + (size_t)h * W_ + w0;
  const float* in2b = in2 + (size_t)b * CHW_;

  v8f acc1 = {};  // columns n = 0..15  (w2 = w0-4+n)
  v8f acc2 = {};  // columns n = 8..23  (w2 = w0+4+n')

  for (int kb = 0; kb < C_; kb += 32) {
    __syncthreads();
    // ---- stage in1 tile: (k,k+1) x (m,m+1) quads; b64 loads, b64 LDS stores ----
    if (tid < 128) {
      int p = tid >> 3;            // channel pair 0..15
      int m = (tid & 7) * 2;       // column pair 0..14
      const float* g = in1b + (size_t)(kb + 2 * p) * HW_ + m;
      float2 x0 = *reinterpret_cast<const float2*>(g);        // channel k,   cols m, m+1
      float2 x1 = *reinterpret_cast<const float2*>(g + HW_);  // channel k+1, cols m, m+1
      unsigned hw0, lw0, hw1, lw1;
      split_pack(x0.x, x1.x, hw0, lw0);
      split_pack(x0.y, x1.y, hw1, lw1);
      *reinterpret_cast<uint2*>(&s1h[p * A_S + m]) = make_uint2(hw0, hw1);
      *reinterpret_cast<uint2*>(&s1l[p * A_S + m]) = make_uint2(lw0, lw1);
    }
    // ---- stage in2: 9 rows (h+di-4) x 32k x 24 cols (w0-4..w0+19), zero padded ----
#pragma unroll
    for (int it = 0; it < 6; ++it) {
      int idx = it * 288 + tid;            // 0..1727 = 9*16*12 quads
      int n   = (idx % 12) * 2;            // column pair (even)
      int p   = (idx / 12) & 15;           // channel pair
      int di  = idx / (12 * 16);
      int h2  = h + di - MAXD;
      int w2  = w0 - MAXD + n;             // even; pairs never straddle the W boundary
      float2 x0 = make_float2(0.f, 0.f), x1 = make_float2(0.f, 0.f);
      if ((unsigned)h2 < (unsigned)H_ && (unsigned)w2 < (unsigned)W_) {
        const float* g = in2b + (size_t)(kb + 2 * p) * HW_ + (size_t)h2 * W_ + w2;
        x0 = *reinterpret_cast<const float2*>(g);
        x1 = *reinterpret_cast<const float2*>(g + HW_);
      }
      unsigned hw0, lw0, hw1, lw1;
      split_pack(x0.x, x1.x, hw0, lw0);
      split_pack(x0.y, x1.y, hw1, lw1);
      int off = (di * 16 + p) * B_S + n;
      *reinterpret_cast<uint2*>(&s2h[off]) = make_uint2(hw0, hw1);
      *reinterpret_cast<uint2*>(&s2l[off]) = make_uint2(lw0, lw1);
    }
    __syncthreads();
    // ---- wave 'wv' computes its di row: 2 band tiles x 3 split terms, interleaved ----
    v16bf aH = load_a_frag(s1h, mn, sel);
    v16bf aL = load_a_frag(s1l, mn, sel);
    int base = wv * 16 * B_S;
    v16bf b1H = load_b_frag(s2h, base, mn,     sel);
    v16bf b1L = load_b_frag(s2l, base, mn,     sel);
    v16bf b2H = load_b_frag(s2h, base, mn + 8, sel);
    v16bf b2L = load_b_frag(s2l, base, mn + 8, sel);
    acc1 = __builtin_amdgcn_wmma_f32_16x16x32_bf16(false, aH, false, b1H, (short)0, acc1, false, false);
    acc2 = __builtin_amdgcn_wmma_f32_16x16x32_bf16(false, aH, false, b2H, (short)0, acc2, false, false);
    acc1 = __builtin_amdgcn_wmma_f32_16x16x32_bf16(false, aH, false, b1L, (short)0, acc1, false, false);
    acc2 = __builtin_amdgcn_wmma_f32_16x16x32_bf16(false, aH, false, b2L, (short)0, acc2, false, false);
    acc1 = __builtin_amdgcn_wmma_f32_16x16x32_bf16(false, aL, false, b1H, (short)0, acc1, false, false);
    acc2 = __builtin_amdgcn_wmma_f32_16x16x32_bf16(false, aL, false, b2H, (short)0, acc2, false, false);
  }

  // ---- spill C/D tiles to LDS, gather band diagonal, coalesced store ----
  __syncthreads();
  {
    float* s = scr + wv * 512;
#pragma unroll
    for (int r = 0; r < 8; ++r) {
      s[r * 32 + lane]       = acc1[r];
      s[256 + r * 32 + lane] = acc2[r];
    }
  }
  __syncthreads();
  {
    const int di = wv;
    float* ob = out + ((size_t)(b * 81 + di * 9) * H_ + h) * W_ + w0;
#pragma unroll
    for (int e = 0; e < 5; ++e) {
      int idx = e * 32 + lane;             // 144 outputs per wave
      if (idx < 144) {
        int m  = idx & 15, dj = idx >> 4;
        int t  = m >> 3;                   // tile select (m<8 -> tile1)
        int c  = m + dj - (t ? 8 : 0);     // column within tile, 0..15
        int r  = m & 7;                    // C/D VGPR row
        float v = scr[wv * 512 + t * 256 + r * 32 + (c + (t ? 16 : 0))];
        ob[(size_t)dj * HW_ + m] = v;
      }
    }
  }
}

extern "C" void kernel_launch(void* const* d_in, const int* in_sizes, int n_in,
                              void* d_out, int out_size, void* d_ws, size_t ws_size,
                              hipStream_t stream) {
  (void)in_sizes; (void)n_in; (void)d_ws; (void)ws_size; (void)out_size;
  const float* in1 = (const float*)d_in[0];
  const float* in2 = (const float*)d_in[1];
  float* out = (float*)d_out;
  dim3 grid(W_ / 16, H_, B_);   // (12, 96, 8) = 9216 workgroups
  dim3 block(288);              // 9 wave32s; wave i owns displacement row di=i
  IterSpatialCorrelationSampler_21560735826388_kernel<<<grid, block, 0, stream>>>(in1, in2, out);
}
